// HeterogeneousNeuralTreeNetwork_25718264169194
// MI455X (gfx1250) — matrix operand order repaired and checked by reading, not compiled, not run on hardware
//
#include <hip/hip_runtime.h>
#include <hip/hip_bf16.h>

// ---------------- problem sizes (match reference) ----------------
#define N_OBJ   100000
#define N_ROOM  10000
#define N_OR    120000
#define N_RR    15000
#define N_OV    100000
#define N_RV    10000
#define E_OV_OR 300000
#define E_RV_RR 45000
#define E_O_OR  300000
#define E_R_OR  240000
#define E_R_RR  45000
#define D_IN    300
#define HID     64
#define OUT_D   25
#define NEG_SLOPE 0.2f

typedef __attribute__((ext_vector_type(16))) _Float16 v16h;
typedef __attribute__((ext_vector_type(8)))  float    v8f;

static inline long cdivl(long a, long b) { return (a + b - 1) / b; }

// ================= device helpers =================
__device__ __forceinline__ unsigned fenc(float f) {
  unsigned u = __float_as_uint(f);
  return (u & 0x80000000u) ? ~u : (u | 0x80000000u);
}
__device__ __forceinline__ float fdec(unsigned e) {
  return __uint_as_float((e & 0x80000000u) ? (e ^ 0x80000000u) : ~e);
}
__device__ __forceinline__ unsigned pack2h(float a, float b) {
  union { _Float16 h[2]; unsigned u; } x;
  x.h[0] = (_Float16)a; x.h[1] = (_Float16)b;
  return x.u;
}

// ================= elementwise / fill =================
__global__ void k_fill_f32(float* __restrict__ p, float v, long n) {
  long i = (long)blockIdx.x * blockDim.x + threadIdx.x;
  long st = (long)gridDim.x * blockDim.x;
  for (; i < n; i += st) p[i] = v;
}
__global__ void k_fill_u32(unsigned* __restrict__ p, unsigned v, long n) {
  long i = (long)blockIdx.x * blockDim.x + threadIdx.x;
  long st = (long)gridDim.x * blockDim.x;
  for (; i < n; i += st) p[i] = v;
}
__global__ void k_scale_relu(float* __restrict__ p, float s, int relu, long n) {
  long i = (long)blockIdx.x * blockDim.x + threadIdx.x;
  long st = (long)gridDim.x * blockDim.x;
  for (; i < n; i += st) {
    float v = p[i] * s;
    p[i] = relu ? fmaxf(v, 0.0f) : v;
  }
}
template <int D>
__global__ void k_bias_init(float* __restrict__ out, const float* __restrict__ b, int N) {
  long i = (long)blockIdx.x * blockDim.x + threadIdx.x;
  long st = (long)gridDim.x * blockDim.x;
  long n = (long)N * D;
  for (; i < n; i += st) out[i] = b[(int)(i % D)];
}

// ================= WMMA f16 GEMM (LDS-staged, branchless edges) =================
// C[M,N] = (flags&1 ? C : 0) + (flags&2 ? bias : 0) + A[M,K] * B[K,N]
// Block tile 64x64, 128 threads (4 waves). Wave w owns rows [bm+16w, bm+16w+16),
// 4 x WMMA 16x16x32 per K-step. A/B staged in LDS as f16; interior blocks use
// unguarded (vectorizable) loads; edge tiles use clamped addresses + value
// selects, so there is NO per-element exec-mask control flow anywhere.
// All fragments (A + 4xB) are loaded into registers BEFORE the WMMA burst so the
// hardware waits once on DScnt and issues the 4 WMMAs back-to-back.
#define BM 64
#define BN 64
#define BK 32
#define LDSTR 40   // padded stride in halves: fragments 16B-aligned, bank-conflict-free

__global__ void __launch_bounds__(128)
k_gemm_wmma_f16(const float* __restrict__ A, const float* __restrict__ B,
                float* __restrict__ C, const float* __restrict__ bias,
                int M, int N, int K, int flags) {
  __shared__ _Float16 sA[BM * LDSTR];   // [row][k]  5120 B
  __shared__ _Float16 sB[BN * LDSTR];   // [col][k]  5120 B (B staged transposed)

  const int tid  = threadIdx.x;
  const int lane = tid & 31;
  const int wv   = tid >> 5;            // wave 0..3
  const int ln   = lane & 15;
  const int kh   = lane >> 4;           // K-half owned by this lane

  const int tilesN = (N + BN - 1) / BN;
  const int bm = (int)(blockIdx.x / tilesN) * BM;
  const int bn = (int)(blockIdx.x % tilesN) * BN;
  const int rowBase = bm + wv * 16;

  const bool intM = (bm + BM <= M);     // uniform interior tests
  const bool intN = (bn + BN <= N);
  const bool evenK = (K & 1) == 0;

  // ---- accumulator init (accumulate / bias) ----
  v8f acc[4];
#pragma unroll
  for (int t = 0; t < 4; ++t)
#pragma unroll
    for (int r = 0; r < 8; ++r) acc[t][r] = 0.0f;

  if (flags & 1) {
    if (intM && intN) {
#pragma unroll
      for (int t = 0; t < 4; ++t)
#pragma unroll
        for (int r = 0; r < 8; ++r)
          acc[t][r] = C[(long)(rowBase + r + 8 * kh) * N + bn + t * 16 + ln];
    } else {
#pragma unroll
      for (int t = 0; t < 4; ++t)
#pragma unroll
        for (int r = 0; r < 8; ++r) {
          int row = rowBase + r + 8 * kh, col = bn + t * 16 + ln;
          float cv = C[(long)min(row, M - 1) * N + min(col, N - 1)];
          acc[t][r] = (row < M && col < N) ? cv : 0.0f;
        }
    }
  }
  if (flags & 2) {
#pragma unroll
    for (int t = 0; t < 4; ++t) {
      int col = bn + t * 16 + ln;
      float bv = bias[min(col, N - 1)];
      bv = (col < N) ? bv : 0.0f;
#pragma unroll
      for (int r = 0; r < 8; ++r) acc[t][r] += bv;
    }
  }

  unsigned* sA32 = (unsigned*)sA;
  unsigned* sB32 = (unsigned*)sB;

  for (int k0 = 0; k0 < K; k0 += BK) {
    const bool kfull = (k0 + BK <= K);
    __syncthreads();
    // ---- stage A: 64x32 f32 -> f16 pairs (1024 dwords, 8 per thread) ----
    if (kfull && intM && evenK) {
#pragma unroll
      for (int p = tid; p < BM * BK / 2; p += 128) {
        int row = p >> 4, kp = p & 15;
        const float2 f = *(const float2*)(A + (long)(bm + row) * K + k0 + 2 * kp);
        sA32[row * (LDSTR / 2) + kp] = pack2h(f.x, f.y);
      }
    } else {
#pragma unroll
      for (int p = tid; p < BM * BK / 2; p += 128) {
        int row = p >> 4, kp = p & 15;
        int k = k0 + 2 * kp, gr = bm + row;
        long base = (long)min(gr, M - 1) * K;
        float f0 = A[base + min(k, K - 1)];
        float f1 = A[base + min(k + 1, K - 1)];
        f0 = (gr < M && k < K) ? f0 : 0.0f;
        f1 = (gr < M && k + 1 < K) ? f1 : 0.0f;
        sA32[row * (LDSTR / 2) + kp] = pack2h(f0, f1);
      }
    }
    // ---- stage B transposed: sB[col][k] (coalesced over cols) ----
    if (kfull && intN) {
#pragma unroll
      for (int p = tid; p < BN * BK / 2; p += 128) {
        int col = p & 63, kp = p >> 6;
        const float* bp = B + (long)(k0 + 2 * kp) * N + bn + col;
        sB32[col * (LDSTR / 2) + kp] = pack2h(bp[0], bp[N]);
      }
    } else {
#pragma unroll
      for (int p = tid; p < BN * BK / 2; p += 128) {
        int col = p & 63, kp = p >> 6;
        int k = k0 + 2 * kp, gc = bn + col;
        int gcc = min(gc, N - 1);
        float f0 = B[(long)min(k, K - 1) * N + gcc];
        float f1 = B[(long)min(k + 1, K - 1) * N + gcc];
        f0 = (gc < N && k < K) ? f0 : 0.0f;
        f1 = (gc < N && k + 1 < K) ? f1 : 0.0f;
        sB32[col * (LDSTR / 2) + kp] = pack2h(f0, f1);
      }
    }
    __syncthreads();

    // ---- fragments from LDS: contiguous 16B chunks, no guards ----
    // A 16x32 f16 layout: lane half kh owns K kh*16..kh*16+15
    v16h af;
    const _Float16* rp = &sA[(wv * 16 + ln) * LDSTR];
#pragma unroll
    for (int j = 0; j < 16; ++j) af[j] = rp[kh * 16 + j];
    // B 32x16 f16 layout: lanes 0-15 hold K=0..15 of their column, lanes 16-31 K=16..31
    v16h bf[4];
#pragma unroll
    for (int t = 0; t < 4; ++t) {
      const _Float16* cp = &sB[(t * 16 + ln) * LDSTR];
#pragma unroll
      for (int j = 0; j < 16; ++j) bf[t][j] = cp[kh * 16 + j];
    }
    // ---- 4 independent WMMAs back-to-back (single DScnt wait before burst) ----
#pragma unroll
    for (int t = 0; t < 4; ++t)
      acc[t] = __builtin_amdgcn_wmma_f32_16x16x32_f16(false, af, false, bf[t],
                                                      (short)0, acc[t], false, false);
  }

  // ---- store ----
  if (intM && intN) {
#pragma unroll
    for (int t = 0; t < 4; ++t)
#pragma unroll
      for (int r = 0; r < 8; ++r)
        C[(long)(rowBase + r + 8 * kh) * N + bn + t * 16 + ln] = acc[t][r];
  } else {
#pragma unroll
    for (int t = 0; t < 4; ++t)
#pragma unroll
      for (int r = 0; r < 8; ++r) {
        int row = rowBase + r + 8 * kh, col = bn + t * 16 + ln;
        if (row < M && col < N) C[(long)row * N + col] = acc[t][r];
      }
  }
}

// ================= matvec helpers =================
// out[i] = dot(A[i,:], v) : one wave32 per row, coalesced lane-strided reads
__global__ void k_matvec(const float* __restrict__ A, const float* __restrict__ v,
                         float* __restrict__ out, int M, int K) {
  int i = blockIdx.x * (blockDim.x >> 5) + (threadIdx.x >> 5);
  int lane = threadIdx.x & 31;
  if (i >= M) return;
  const float* row = A + (long)i * K;
  float s = 0.0f;
  for (int k = lane; k < K; k += 32) s += row[k] * v[k];
#pragma unroll
  for (int off = 16; off > 0; off >>= 1) s += __shfl_down(s, off, 32);
  if (lane == 0) out[i] = s;
}
// out[k] = dot(W[k,:], a)   (folds Wd @ a_d for the GAT destination projection)
__global__ void k_w_times_a(const float* __restrict__ W, const float* __restrict__ a,
                            float* __restrict__ out, int K, int N) {
  int k = blockIdx.x * blockDim.x + threadIdx.x;
  if (k >= K) return;
  const float* row = W + (long)k * N;
  float s = 0.0f;
  for (int n = 0; n < N; ++n) s += row[n] * a[n];
  out[k] = s;
}

// ================= segment / scatter kernels (D compile-time: no divisions) =================
__global__ void k_edge_logit_max(const int* __restrict__ src, const int* __restrict__ dst,
                                 const float* __restrict__ as_, const float* __restrict__ ad_,
                                 float* __restrict__ logit, unsigned* __restrict__ mEnc, int E) {
  int e = blockIdx.x * blockDim.x + threadIdx.x;
  if (e >= E) return;
  float x = as_[src[e]] + ad_[dst[e]];
  x = (x > 0.0f) ? x : NEG_SLOPE * x;          // leaky_relu
  logit[e] = x;
  atomicMax(&mEnc[dst[e]], fenc(x));           // segment max (encoded-uint float max)
}
__global__ void k_edge_expsum(const int* __restrict__ dst, float* __restrict__ ew,
                              const unsigned* __restrict__ mEnc, float* __restrict__ ssum, int E) {
  int e = blockIdx.x * blockDim.x + threadIdx.x;
  if (e >= E) return;
  int d = dst[e];
  float w = __expf(ew[e] - fdec(mEnc[d]));     // ew holds logits on entry
  ew[e] = w;
  atomicAdd(&ssum[d], w);
}
template <int D>
__global__ void k_gat_scatter(const int* __restrict__ src, const int* __restrict__ dst,
                              const float* __restrict__ hs, const float* __restrict__ ew,
                              const float* __restrict__ ssum, float* __restrict__ out, int E) {
  long i = (long)blockIdx.x * blockDim.x + threadIdx.x;
  long st = (long)gridDim.x * blockDim.x;
  long n = (long)E * D;
  for (; i < n; i += st) {
    int e = (int)(i / D), d = (int)(i % D);
    int dd = dst[e];
    float w = ew[e] / (ssum[dd] + 1e-16f);
    atomicAdd(&out[(long)dd * D + d], hs[(long)src[e] * D + d] * w);
  }
}
template <int D>
__global__ void k_scatter_add(const int* __restrict__ src, const int* __restrict__ dst,
                              const float* __restrict__ x, float* __restrict__ agg, int E) {
  long i = (long)blockIdx.x * blockDim.x + threadIdx.x;
  long st = (long)gridDim.x * blockDim.x;
  long n = (long)E * D;
  for (; i < n; i += st) {
    int e = (int)(i / D), d = (int)(i % D);
    atomicAdd(&agg[(long)dst[e] * D + d], x[(long)src[e] * D + d]);
  }
}
__global__ void k_count(const int* __restrict__ dst, float* __restrict__ cnt, int E) {
  int e = blockIdx.x * blockDim.x + threadIdx.x;
  if (e >= E) return;
  atomicAdd(&cnt[dst[e]], 1.0f);
}
template <int D>
__global__ void k_mean_inplace(float* __restrict__ agg, const float* __restrict__ cnt, int N) {
  long i = (long)blockIdx.x * blockDim.x + threadIdx.x;
  long st = (long)gridDim.x * blockDim.x;
  long n = (long)N * D;
  for (; i < n; i += st) agg[i] /= fmaxf(cnt[i / D], 1.0f);
}
template <int D>
__global__ void k_mean_out(float* __restrict__ out, const float* __restrict__ agg,
                           const float* __restrict__ cnt, int N) {
  long i = (long)blockIdx.x * blockDim.x + threadIdx.x;
  long st = (long)gridDim.x * blockDim.x;
  long n = (long)N * D;
  for (; i < n; i += st) out[i] = agg[i] / fmaxf(cnt[i / D], 1.0f);
}

// ================= host-side helpers =================
static void launch_fill(float* p, float v, long n, hipStream_t s) {
  k_fill_f32<<<(unsigned)cdivl(n, 256), 256, 0, s>>>(p, v, n);
}
static void launch_gemm(const float* A, const float* B, float* C, const float* bias,
                        int M, int N, int K, int flags, hipStream_t s) {
  long blocks = cdivl(M, BM) * cdivl(N, BN);
  k_gemm_wmma_f16<<<(unsigned)blocks, 128, 0, s>>>(A, B, C, bias, M, N, K, flags);
}
static void launch_scatter_add(const int* src, const int* dst, const float* x, float* agg,
                               int E, int D, hipStream_t s) {
  unsigned g = (unsigned)cdivl((long)E * D, 256);
  if (D == 300)      k_scatter_add<300><<<g, 256, 0, s>>>(src, dst, x, agg, E);
  else if (D == 64)  k_scatter_add<64><<<g, 256, 0, s>>>(src, dst, x, agg, E);
  else               k_scatter_add<25><<<g, 256, 0, s>>>(src, dst, x, agg, E);
}
static void launch_mean_inplace(float* agg, const float* cnt, int N, int D, hipStream_t s) {
  unsigned g = (unsigned)cdivl((long)N * D, 256);
  if (D == 300)      k_mean_inplace<300><<<g, 256, 0, s>>>(agg, cnt, N);
  else if (D == 64)  k_mean_inplace<64><<<g, 256, 0, s>>>(agg, cnt, N);
  else               k_mean_inplace<25><<<g, 256, 0, s>>>(agg, cnt, N);
}

// SAGEConv(mean): out (+)= mean_j(xs[j]) @ Wl + bl + xt @ Wr
static void run_sage(const float* xs, const int* ei, int E,
                     const float* xt, int n_dst, int din, int dout,
                     const float* Wl, const float* Wr, const float* bl,
                     float* out, bool accum, float* agg, float* cnt, hipStream_t s) {
  const int* src = ei;
  const int* dst = ei + E;
  launch_fill(agg, 0.0f, (long)n_dst * din, s);
  launch_fill(cnt, 0.0f, n_dst, s);
  launch_scatter_add(src, dst, xs, agg, E, din, s);
  k_count<<<(unsigned)cdivl(E, 256), 256, 0, s>>>(dst, cnt, E);
  launch_mean_inplace(agg, cnt, n_dst, din, s);
  launch_gemm(agg, Wl, out, bl, n_dst, dout, din, (accum ? 1 : 0) | 2, s);  // mean@Wl + bl (+C)
  launch_gemm(xt, Wr, out, nullptr, n_dst, dout, din, 1, s);                // += xt@Wr
}

// GATConv((ds,dt),dt,heads=1): out = segsum(softmax_dst(lrelu(as[src]+ad[dst])) * hs[src]) + b
// ht is only consumed as ht@a_d, folded to xdst @ (Wd@a_d)  -> saves an N_OR x 300 x 300 GEMM.
static void run_gat(const float* xsrc, int n_src, const float* xdst, int n_dst,
                    const int* ei, int E,
                    const float* Wd, const float* Ws, const float* a_d, const float* a_s,
                    const float* b, float* out, float* hs, float* alpha_s, float* alpha_d,
                    unsigned* mEnc, float* ssum, float* ew, float* wv, hipStream_t s) {
  const int* src = ei;
  const int* dst = ei + E;
  launch_gemm(xsrc, Ws, hs, nullptr, n_src, D_IN, D_IN, 0, s);              // hs = xs @ Ws (WMMA)
  k_matvec<<<(unsigned)cdivl(n_src, 8), 256, 0, s>>>(hs, a_s, alpha_s, n_src, D_IN);
  k_w_times_a<<<(unsigned)cdivl(D_IN, 256), 256, 0, s>>>(Wd, a_d, wv, D_IN, D_IN);
  k_matvec<<<(unsigned)cdivl(n_dst, 8), 256, 0, s>>>(xdst, wv, alpha_d, n_dst, D_IN);
  k_fill_u32<<<(unsigned)cdivl(n_dst, 256), 256, 0, s>>>(mEnc, 0u, n_dst);
  launch_fill(ssum, 0.0f, n_dst, s);
  k_edge_logit_max<<<(unsigned)cdivl(E, 256), 256, 0, s>>>(src, dst, alpha_s, alpha_d, ew, mEnc, E);
  k_edge_expsum<<<(unsigned)cdivl(E, 256), 256, 0, s>>>(dst, ew, mEnc, ssum, E);
  k_bias_init<D_IN><<<(unsigned)cdivl((long)n_dst * D_IN, 256), 256, 0, s>>>(out, b, n_dst);
  k_gat_scatter<D_IN><<<(unsigned)cdivl((long)E * D_IN, 256), 256, 0, s>>>(src, dst, hs, ew, ssum, out, E);
}

// ================= entry point =================
extern "C" void kernel_launch(void* const* d_in, const int* in_sizes, int n_in,
                              void* d_out, int out_size, void* d_ws, size_t ws_size,
                              hipStream_t stream) {
  (void)in_sizes; (void)n_in; (void)out_size; (void)ws_size;

  // ---- inputs (setup_inputs insertion order; params pytree-flattened, sorted keys) ----
  const float* x_object = (const float*)d_in[0];
  const float* x_room   = (const float*)d_in[1];
  const float* x_or_in  = (const float*)d_in[2];
  const float* x_rr_in  = (const float*)d_in[3];
  const float* x_ov     = (const float*)d_in[4];
  const float* x_rv     = (const float*)d_in[5];
  const int* ei_ov_or = (const int*)d_in[6];
  const int* ei_rv_rr = (const int*)d_in[7];
  const int* ei_o_or  = (const int*)d_in[8];
  const int* ei_or_o  = (const int*)d_in[9];
  const int* ei_r_or  = (const int*)d_in[10];
  const int* ei_or_r  = (const int*)d_in[11];
  const int* ei_r_rr  = (const int*)d_in[12];
  const int* ei_rr_r  = (const int*)d_in[13];
  const int* ei_r_rv  = (const int*)d_in[14];

  // params["layers"][L][conv in sorted order {o_or, or_o, or_r, r_or, r_rr, rr_r}] -> [Wl, Wr, bl]
  enum { C_O_OR = 0, C_OR_O = 1, C_OR_R = 2, C_R_OR = 3, C_R_RR = 4, C_RR_R = 5 };
#define LP(L, CONV, W) ((const float*)d_in[15 + (L) * 18 + (CONV) * 3 + (W)])
  // params["pre"] {ov_or, rv_rr} -> [Wd, Ws, a_d, a_s, b], base index 15 + 4*18 = 87
#define GP(G, W) ((const float*)d_in[87 + (G) * 5 + (W)])

  // ---- workspace layout (floats) ----
  float* W = (float*)d_ws;
  size_t off = 0;
  auto take = [&](size_t n) { size_t r = off; off += (n + 63) & ~(size_t)63; return r; };
  const size_t XOR_A  = take((size_t)N_OR  * D_IN);
  const size_t XRR_A  = take((size_t)N_RR  * D_IN);
  const size_t XOR_B  = take((size_t)N_OR  * HID);
  const size_t XRR_B  = take((size_t)N_RR  * HID);
  const size_t OBJ_A  = take((size_t)N_OBJ * HID);
  const size_t OBJ_B  = take((size_t)N_OBJ * HID);
  const size_t ROOM_A = take((size_t)N_ROOM * HID);
  const size_t ROOM_B = take((size_t)N_ROOM * HID);
  const size_t HS     = take((size_t)N_OV  * D_IN);
  const size_t AGG    = take((size_t)N_OR  * D_IN);   // max n_dst * max din
  const size_t CNT    = take(N_OR);
  const size_t AS     = take(N_OV);
  const size_t AD     = take(N_OR);
  const size_t MENC   = take(N_OR);
  const size_t SSUM   = take(N_OR);
  const size_t EW     = take(E_OV_OR);
  const size_t WV     = take(512);

  float* agg = W + AGG;  float* cnt = W + CNT;

  // ---- pre_mp: GATConvs initialize clique node features ----
  run_gat(x_ov, N_OV, x_or_in, N_OR, ei_ov_or, E_OV_OR,
          GP(0, 0), GP(0, 1), GP(0, 2), GP(0, 3), GP(0, 4),
          W + XOR_A, W + HS, W + AS, W + AD, (unsigned*)(W + MENC), W + SSUM, W + EW, W + WV,
          stream);
  run_gat(x_rv, N_RV, x_rr_in, N_RR, ei_rv_rr, E_RV_RR,
          GP(1, 0), GP(1, 1), GP(1, 2), GP(1, 3), GP(1, 4),
          W + XRR_A, W + HS, W + AS, W + AD, (unsigned*)(W + MENC), W + SSUM, W + EW, W + WV,
          stream);

  // ---- 4 layers of HeteroConv(SAGEConv, aggr='mean' across edge types) ----
  const float* cur_o  = x_object;
  const float* cur_r  = x_room;
  const float* cur_or = W + XOR_A;
  const float* cur_rr = W + XRR_A;
  const int dims[5] = {D_IN, HID, HID, HID, OUT_D};

  for (int L = 0; L < 4; ++L) {
    int din = dims[L], dout = dims[L + 1];
    float* o_out  = W + ((L & 1) ? OBJ_A  : OBJ_B);
    float* r_out  = W + ((L & 1) ? ROOM_A : ROOM_B);
    float* or_out = W + ((L & 1) ? XOR_A  : XOR_B);
    float* rr_out = W + ((L & 1) ? XRR_A  : XRR_B);

    // object <- or
    run_sage(cur_or, ei_or_o, E_O_OR, cur_o, N_OBJ, din, dout,
             LP(L, C_OR_O, 0), LP(L, C_OR_O, 1), LP(L, C_OR_O, 2), o_out, false, agg, cnt, stream);
    // room <- 0.5*(or + rr)
    run_sage(cur_or, ei_or_r, E_R_OR, cur_r, N_ROOM, din, dout,
             LP(L, C_OR_R, 0), LP(L, C_OR_R, 1), LP(L, C_OR_R, 2), r_out, false, agg, cnt, stream);
    run_sage(cur_rr, ei_rr_r, E_R_RR, cur_r, N_ROOM, din, dout,
             LP(L, C_RR_R, 0), LP(L, C_RR_R, 1), LP(L, C_RR_R, 2), r_out, true, agg, cnt, stream);
    // or <- 0.5*(object + room)
    run_sage(cur_o, ei_o_or, E_O_OR, cur_or, N_OR, din, dout,
             LP(L, C_O_OR, 0), LP(L, C_O_OR, 1), LP(L, C_O_OR, 2), or_out, false, agg, cnt, stream);
    run_sage(cur_r, ei_r_or, E_R_OR, cur_or, N_OR, din, dout,
             LP(L, C_R_OR, 0), LP(L, C_R_OR, 1), LP(L, C_R_OR, 2), or_out, true, agg, cnt, stream);
    // rr <- room
    run_sage(cur_r, ei_r_rr, E_R_RR, cur_rr, N_RR, din, dout,
             LP(L, C_R_RR, 0), LP(L, C_R_RR, 1), LP(L, C_R_RR, 2), rr_out, false, agg, cnt, stream);

    int relu = (L != 3) ? 1 : 0;
    k_scale_relu<<<(unsigned)cdivl((long)N_OBJ * dout, 256), 256, 0, stream>>>(o_out, 1.0f, relu, (long)N_OBJ * dout);
    k_scale_relu<<<(unsigned)cdivl((long)N_ROOM * dout, 256), 256, 0, stream>>>(r_out, 0.5f, relu, (long)N_ROOM * dout);
    k_scale_relu<<<(unsigned)cdivl((long)N_OR * dout, 256), 256, 0, stream>>>(or_out, 0.5f, relu, (long)N_OR * dout);
    k_scale_relu<<<(unsigned)cdivl((long)N_RR * dout, 256), 256, 0, stream>>>(rr_out, 1.0f, relu, (long)N_RR * dout);

    cur_o = o_out; cur_r = r_out; cur_or = or_out; cur_rr = rr_out;
  }

  // ---- LeafPool(mean): room -> room_virtual, written straight to d_out ----
  float* out = (float*)d_out;
  launch_fill(agg, 0.0f, (long)N_RV * OUT_D, stream);
  launch_fill(cnt, 0.0f, N_RV, stream);
  launch_scatter_add(ei_r_rv, ei_r_rv + N_ROOM, cur_r, agg, N_ROOM, OUT_D, stream);
  k_count<<<(unsigned)cdivl(N_ROOM, 256), 256, 0, stream>>>(ei_r_rv + N_ROOM, cnt, N_ROOM);
  k_mean_out<OUT_D><<<(unsigned)cdivl((long)N_RV * OUT_D, 256), 256, 0, stream>>>(out, agg, cnt, N_RV);
}
